// FrequencyBandModulation_63780264346253
// MI455X (gfx1250) — matrix operand order repaired and checked by reading, compiled, not run on hardware
//
#include <hip/hip_runtime.h>
#include <math.h>

// ---------------------------------------------------------------------------
// FrequencyBandModulation on MI455X (gfx1250, wave32, WMMA)
//
// low_k = MhRe_k . x . Dr_k^T  -  MhIm_k . x . Di_k^T   (per 128x128 image)
// gate_k = 2*sigmoid(conv3x3_{256->1}(x) + b_k)         (tap-GEMM + shifts)
// out = sum_k gate_k * (pre_{k-1} - low_k) + low_last
// ---------------------------------------------------------------------------

typedef __attribute__((ext_vector_type(16))) __bf16 v16bf;
typedef __attribute__((ext_vector_type(8)))  __bf16 bf16x8;
typedef __attribute__((ext_vector_type(8)))  float  v8f;

#define HW     128
#define NPIX   16384      // 128*128
#define CH     256
#define BATCH  8
#define NBAND  3
#define LDSP   136        // 128 + 8 halves pad (16B-aligned rows, bank stagger)

// workspace layout (bytes), all offsets 256B aligned
#define WS_MHRE   0u          // 3*128*128 bf16  =  98304
#define WS_MHIM   98304u      //                  98304
#define WS_BW     196608u     // 3*128*256 bf16  = 196608  (Bw[n][kk]: kk<128 -> Dr, kk>=128 -> -Di)
#define WS_BC     393216u     // 32*256 bf16     =  16384  (conv taps, transposed, padded to 32)
#define WS_YTAP   409600u     // 131072*32 f32   = 16777216
#define WS_GATES  17186816u   // 8*3*16384 f32   = 1572864
// total ~18.8 MB

static __device__ __forceinline__ v16bf cat16(bf16x8 lo, bf16x8 hi) {
  return __builtin_shufflevector(lo, hi, 0,1,2,3,4,5,6,7,8,9,10,11,12,13,14,15);
}
// A-operand (16x32 bf16): lane m = lane&15, K chunks at k0+sel*8 and k0+16+sel*8
static __device__ __forceinline__ v16bf ldA(const __bf16* row, int k0, int sel) {
  bf16x8 lo = *(const bf16x8*)(row + k0 + sel * 8);
  bf16x8 hi = *(const bf16x8*)(row + k0 + 16 + sel * 8);
  return cat16(lo, hi);
}
// B-operand (32x16 bf16), stored as B^T rows: K chunks at k0+sel*8 and k0+16+sel*8
static __device__ __forceinline__ v16bf ldB(const __bf16* rowT, int k0, int sel) {
  bf16x8 lo = *(const bf16x8*)(rowT + k0 + sel * 8);
  bf16x8 hi = *(const bf16x8*)(rowT + k0 + 16 + sel * 8);
  return cat16(lo, hi);
}

// ---------------------------------------------------------------------------
// Kernel 1: build band operator matrices + repacked conv weights (bf16)
// ---------------------------------------------------------------------------
__global__ __launch_bounds__(256) void fbm_precompute(const float* __restrict__ W,
                                                      char* __restrict__ ws) {
  __bf16* MhRe = (__bf16*)(ws + WS_MHRE);
  __bf16* MhIm = (__bf16*)(ws + WS_MHIM);
  __bf16* Bw   = (__bf16*)(ws + WS_BW);
  __bf16* Bc   = (__bf16*)(ws + WS_BC);
  const float step = 6.28318530717958647692f / 128.0f;
  const int stride = gridDim.x * blockDim.x;
  const int tid0 = blockIdx.x * blockDim.x + threadIdx.x;

  // Mh (complex circulant from asymmetric h-mask: keep k<mi and k>=64)
  for (int idx = tid0; idx < NBAND * 128 * 128; idx += stride) {
    int band = idx / (128 * 128);
    int r = (idx >> 7) & 127, a = idx & 127;
    int mi = 32 >> band;
    int d = r - a;
    float re = 0.f, im = 0.f;
    for (int k = 0; k < mi; ++k) {
      int e = ((k * d) % 128 + 128) % 128;
      re += cosf(step * e); im += sinf(step * e);
    }
    for (int k = 64; k < 128; ++k) {
      int e = ((k * d) % 128 + 128) % 128;
      re += cosf(step * e); im += sinf(step * e);
    }
    MhRe[idx] = (__bf16)(re * (1.0f / 128.0f));
    MhIm[idx] = (__bf16)(im * (1.0f / 128.0f));
  }
  // Bw[band][n][kk]: kk<128 -> Dr[n-kk], kk>=128 -> -Di[n-(kk-128)]
  for (int idx = tid0; idx < NBAND * 128 * 256; idx += stride) {
    int band = idx / (128 * 256);
    int rem = idx % (128 * 256);
    int n = rem >> 8, kk = rem & 255;
    int mi = 32 >> band;
    float v;
    if (kk < 128) {
      int q = n - kk;
      float s = 1.f;
      for (int j = 1; j < mi; ++j) {
        int e = ((j * q) % 128 + 128) % 128;
        s += 2.f * cosf(step * e);
      }
      v = s * (1.0f / 128.0f);
    } else {
      int q = n - (kk - 128);
      float s = 0.f;
      for (int j = 1; j < mi; ++j) {
        int e = ((j * q) % 128 + 128) % 128;
        s += sinf(step * e);
      }
      v = -s * (2.0f / 128.0f);
    }
    Bw[idx] = (__bf16)v;
  }
  // Bc[t][c], t = band*9 + dy*3+dx (27 used, padded to 32)
  for (int idx = tid0; idx < 32 * 256; idx += stride) {
    int t = idx >> 8, c = idx & 255;
    float v = 0.f;
    if (t < 27) {
      int band = t / 9, r9 = t % 9;
      v = W[(band * 256 + c) * 9 + r9];
    }
    Bc[idx] = (__bf16)v;
  }
}

// ---------------------------------------------------------------------------
// Kernel 2: tap-GEMM  ytap[pixel][32] = x[pixel][C=256] * Bc^T   (WMMA)
// ---------------------------------------------------------------------------
__global__ __launch_bounds__(256, 1) void fbm_conv_taps(const float* __restrict__ x,
                                                        char* __restrict__ ws) {
  const __bf16* Bc = (const __bf16*)(ws + WS_BC);
  float* ytap = (float*)(ws + WS_YTAP);
  const int b  = blockIdx.x >> 7;
  const int p0 = (blockIdx.x & 127) * 128;
  const float* xb = x + (size_t)b * CH * NPIX;

  __shared__ __attribute__((aligned(16))) __bf16 Xc[128][CH + 8];
  const int tid = threadIdx.x;
  for (int i = tid; i < 128 * 256; i += 256) {
    int cc = i >> 7, pp = i & 127;
    Xc[pp][cc] = (__bf16)xb[(size_t)cc * NPIX + p0 + pp];
  }
  __syncthreads();

  const int lane = tid & 31, wv = tid >> 5;
  const int sel = lane >> 4, ln = lane & 15;
#pragma unroll
  for (int t = 0; t < 2; ++t) {
    int id = wv * 2 + t;
    int row0 = (id >> 1) * 16, col0 = (id & 1) * 16;
    v8f acc = {0.f, 0.f, 0.f, 0.f, 0.f, 0.f, 0.f, 0.f};
#pragma unroll
    for (int ks = 0; ks < 8; ++ks) {
      int k0 = ks * 32;
      v16bf a  = ldA(&Xc[row0 + ln][0], k0, sel);
      v16bf bb = ldB(Bc + (col0 + ln) * 256, k0, sel);
      acc = __builtin_amdgcn_wmma_f32_16x16x32_bf16(false, a, false, bb,
                                                    (short)0, acc, false, false);
    }
    float* yt = ytap + ((size_t)b * NPIX + p0 + row0 + sel * 8) * 32 + col0 + ln;
#pragma unroll
    for (int v = 0; v < 8; ++v) yt[(size_t)v * 32] = acc[v];
  }
}

// ---------------------------------------------------------------------------
// Kernel 3: shift-combine taps -> gates = 2*sigmoid(conv + bias)
// ---------------------------------------------------------------------------
__global__ __launch_bounds__(256) void fbm_gates(const float* __restrict__ bias,
                                                 char* __restrict__ ws) {
  const float* ytap = (const float*)(ws + WS_YTAP);
  float* gates = (float*)(ws + WS_GATES);
  int idx = blockIdx.x * blockDim.x + threadIdx.x;
  if (idx >= BATCH * NBAND * NPIX) return;
  int b = idx / (NBAND * NPIX);
  int band = (idx / NPIX) % NBAND;
  int pix = idx & (NPIX - 1);
  int h = pix >> 7, w = pix & 127;
  float sum = bias[band];
#pragma unroll
  for (int dy = 0; dy < 3; ++dy)
#pragma unroll
    for (int dx = 0; dx < 3; ++dx) {
      int hh = h + dy - 1, ww = w + dx - 1;
      if (hh >= 0 && hh < 128 && ww >= 0 && ww < 128)
        sum += ytap[((size_t)b * NPIX + hh * 128 + ww) * 32 + band * 9 + dy * 3 + dx];
    }
  gates[idx] = 2.0f / (1.0f + __expf(-sum));
}

// ---------------------------------------------------------------------------
// Kernel 4: fused band GEMMs + modulation.  One block = one (b,c) image.
// 512 threads = 16 waves, 4 output 16x16 tiles per wave.
// ---------------------------------------------------------------------------
__global__ __launch_bounds__(512, 1) void fbm_main(const float* __restrict__ x,
                                                   const char* __restrict__ ws,
                                                   float* __restrict__ out) {
  const int img = blockIdx.x;                 // b*256 + c
  const int b = img >> 8;
  const float* xi = x + (size_t)img * NPIX;
  float* oi = out + (size_t)img * NPIX;
  const __bf16* MhRe  = (const __bf16*)(ws + WS_MHRE);
  const __bf16* MhIm  = (const __bf16*)(ws + WS_MHIM);
  const __bf16* Bw    = (const __bf16*)(ws + WS_BW);
  const float*  gates = (const float*)(ws + WS_GATES);

  __shared__ __attribute__((aligned(16))) __bf16 XsT[128][LDSP];  // x transposed (B of GEMM1)
  __shared__ __attribute__((aligned(16))) __bf16 Tt[128][LDSP];   // intermediate (A of GEMM2)

  const int tid = threadIdx.x;
  for (int i = tid; i < NPIX; i += 512) XsT[i & 127][i >> 7] = (__bf16)xi[i];

  const int lane = tid & 31, wv = tid >> 5;
  const int sel = lane >> 4, ln = lane & 15;

  int tr[4], tc[4];
  float pre[4][8], outv[4][8];
#pragma unroll
  for (int t = 0; t < 4; ++t) {
    int id = wv * 4 + t;
    tr[t] = id >> 3; tc[t] = id & 7;
    int row0 = tr[t] * 16, col0 = tc[t] * 16;
#pragma unroll
    for (int v = 0; v < 8; ++v) {
      pre[t][v] = xi[(row0 + v + sel * 8) * 128 + col0 + ln];
      outv[t][v] = 0.f;
    }
  }
  __syncthreads();

  for (int band = 0; band < NBAND; ++band) {
    __builtin_prefetch(gates + ((size_t)(b * 3 + band)) * NPIX + (tid << 5), 0, 3);
    v8f low[4];
#pragma unroll
    for (int t = 0; t < 4; ++t) low[t] = (v8f){0.f,0.f,0.f,0.f,0.f,0.f,0.f,0.f};

#pragma unroll 1
    for (int pass = 0; pass < 2; ++pass) {
      const __bf16* Am = (pass ? MhIm : MhRe) + band * NPIX;   // operator rows (L2-resident)
      __syncthreads();                                         // WAR on Tt
      // GEMM1: T = Am . x   (B from transposed-x LDS, contiguous 16B chunks)
#pragma unroll
      for (int t = 0; t < 4; ++t) {
        int row0 = tr[t] * 16, col0 = tc[t] * 16;
        v8f acc = (v8f){0.f,0.f,0.f,0.f,0.f,0.f,0.f,0.f};
#pragma unroll
        for (int ks = 0; ks < 4; ++ks) {
          int k0 = ks * 32;
          v16bf a  = ldA(Am + (row0 + ln) * 128, k0, sel);
          v16bf bb = ldB(&XsT[col0 + ln][0], k0, sel);
          acc = __builtin_amdgcn_wmma_f32_16x16x32_bf16(false, a, false, bb,
                                                        (short)0, acc, false, false);
        }
#pragma unroll
        for (int v = 0; v < 8; ++v)
          Tt[row0 + v + sel * 8][col0 + ln] = (__bf16)acc[v];
      }
      __syncthreads();
      // GEMM2: low += T . Bw_half^T  (Bw stored transposed -> contiguous loads)
      const __bf16* Bwg = Bw + band * 128 * 256 + pass * 128;
#pragma unroll
      for (int t = 0; t < 4; ++t) {
        int row0 = tr[t] * 16, col0 = tc[t] * 16;
        v8f acc = low[t];
#pragma unroll
        for (int ks = 0; ks < 4; ++ks) {
          int k0 = ks * 32;
          v16bf a  = ldA(&Tt[row0 + ln][0], k0, sel);
          v16bf bb = ldB(Bwg + (col0 + ln) * 256, k0, sel);
          acc = __builtin_amdgcn_wmma_f32_16x16x32_bf16(false, a, false, bb,
                                                        (short)0, acc, false, false);
        }
        low[t] = acc;
      }
    }
    // modulation epilogue in WMMA C-layout registers
    const float* gb = gates + ((size_t)(b * 3 + band)) * NPIX;
#pragma unroll
    for (int t = 0; t < 4; ++t) {
      int row0 = tr[t] * 16, col0 = tc[t] * 16;
#pragma unroll
      for (int v = 0; v < 8; ++v) {
        float g = gb[(row0 + v + sel * 8) * 128 + col0 + ln];
        float lo = low[t][v];
        outv[t][v] += g * (pre[t][v] - lo);
        pre[t][v] = lo;
      }
    }
  }
#pragma unroll
  for (int t = 0; t < 4; ++t) {
    int row0 = tr[t] * 16, col0 = tc[t] * 16;
#pragma unroll
    for (int v = 0; v < 8; ++v)
      oi[(row0 + v + sel * 8) * 128 + col0 + ln] = outv[t][v] + pre[t][v];
  }
}

// ---------------------------------------------------------------------------
extern "C" void kernel_launch(void* const* d_in, const int* in_sizes, int n_in,
                              void* d_out, int out_size, void* d_ws, size_t ws_size,
                              hipStream_t stream) {
  const float* x  = (const float*)d_in[0];   // [8,256,128,128]
  const float* W  = (const float*)d_in[1];   // [3,1,256,3,3]
  const float* Bb = (const float*)d_in[2];   // [3,1]
  float* out = (float*)d_out;
  char* ws = (char*)d_ws;                    // ~18.8 MB used

  hipLaunchKernelGGL(fbm_precompute, dim3(256), dim3(256), 0, stream, W, ws);
  hipLaunchKernelGGL(fbm_conv_taps, dim3(BATCH * 128), dim3(256), 0, stream, x, ws);
  int totalG = BATCH * NBAND * NPIX;
  hipLaunchKernelGGL(fbm_gates, dim3((totalG + 255) / 256), dim3(256), 0, stream, Bb, ws);
  hipLaunchKernelGGL(fbm_main, dim3(BATCH * CH), dim3(512), 0, stream, x, ws, out);
}
